// AttentionDIN_1228360647196
// MI455X (gfx1250) — matrix operand order repaired and checked by reading, not compiled, not run on hardware
//
#include <hip/hip_runtime.h>

#define NEG_INF -4294967295.0f

typedef __attribute__((ext_vector_type(16))) __bf16       v16bf;
typedef __attribute__((ext_vector_type(8)))  float        v8f;
typedef __attribute__((ext_vector_type(4)))  unsigned int u32x4;

// Problem dims (fixed by the reference)
constexpr int T  = 200;
constexpr int D  = 64;
constexpr int H1 = 80;
constexpr int H2 = 40;
constexpr int TP = 208;   // padded T = 13 * 16
constexpr int MT = 13;    // M tiles

// LDS row strides (in bf16 halves), padded to dodge bank conflicts
constexpr int KSTR  = 72;    // key panel: 64 + 8
constexpr int W1STR = 264;   // W1^T rows: 256 + 8
constexpr int HSTR  = 104;   // h slab / W2^T rows: 96 + 8

// LDS layout (bytes), all 16B aligned
constexpr int OFF_KBF   = 0;
constexpr int SZ_KBF    = TP * KSTR * 2;            // 29952
constexpr int OFF_W1T   = OFF_KBF + SZ_KBF;
constexpr int SZ_W1T    = H1 * W1STR * 2;           // 42240
constexpr int OFF_W2T   = OFF_W1T + SZ_W1T;
constexpr int SZ_W2T    = 48 * HSTR * 2;            // 9984
constexpr int OFF_HSLAB = OFF_W2T + SZ_W2T;
constexpr int SZ_HSLAB  = 8 * 16 * HSTR * 2;        // 26624
constexpr int OFF_QBF   = OFF_HSLAB + SZ_HSLAB;     // 128 B
constexpr int OFF_W3P   = OFF_QBF + 128;            // 192 B (96 halves, zero padded)
constexpr int OFF_B1    = OFF_W3P + 192;            // 320 B
constexpr int OFF_B2    = OFF_B1 + 320;             // 192 B (48 floats)
constexpr int OFF_SC    = OFF_B2 + 192;             // 832 B (208 floats)
constexpr int OFF_RED   = OFF_SC + 832;             // 32 B
constexpr int OFF_G     = OFF_RED + 32;             // 16 B
constexpr int OFF_PART  = OFF_G + 16;               // 1024 B
constexpr int SMEM_BYTES = OFF_PART + 1024;         // 111536 B

// ---- WMMA fragment helpers -------------------------------------------------
// A-fragment (16x32 bf16): per lane, halves 0..7 hold K = b8+0..7, halves 8..15
// hold K = 16+b8+0..7 (b8 = lane<16 ? 0 : 8)  -> two b128 loads 32B apart.
__device__ __forceinline__ v16bf frag_gap(const char* p) {
  union { v16bf v; u32x4 q[2]; } u;
  u.q[0] = *(const u32x4*)(p);
  u.q[1] = *(const u32x4*)(p + 32);
  return u.v;
}
// B-fragment (32x16 bf16): per lane, halves 0..7 hold K = kb16+0..7, halves 8..15
// hold K = 16+kb16+0..7 (kb16 = lane<16 ? 0 : 8) -> two b128 loads 32B apart.
__device__ __forceinline__ v16bf frag_seq(const char* p) {
  union { v16bf v; u32x4 q[2]; } u;
  u.q[0] = *(const u32x4*)(p);
  u.q[1] = *(const u32x4*)(p + 32);
  return u.v;
}
__device__ __forceinline__ v16bf frag_sub(v16bf a, v16bf b) {
  v16bf r;
#pragma unroll
  for (int i = 0; i < 16; ++i) r[i] = (__bf16)((float)a[i] - (float)b[i]);
  return r;
}
__device__ __forceinline__ v16bf frag_mul(v16bf a, v16bf b) {
  v16bf r;
#pragma unroll
  for (int i = 0; i < 16; ++i) r[i] = (__bf16)((float)a[i] * (float)b[i]);
  return r;
}
__device__ __forceinline__ float sigmoidf_fast(float x) {
  return __fdividef(1.0f, 1.0f + __expf(-x));
}

__global__ __launch_bounds__(256)
void din_attn_kernel(const float* __restrict__ query,
                     const float* __restrict__ key,
                     const int*   __restrict__ mask,
                     const float* __restrict__ W1, const float* __restrict__ b1,
                     const float* __restrict__ W2, const float* __restrict__ b2,
                     const float* __restrict__ W3, const float* __restrict__ b3,
                     float* __restrict__ out)
{
  extern __shared__ char smem[];
  __bf16* kbf = (__bf16*)(smem + OFF_KBF);
  __bf16* w1t = (__bf16*)(smem + OFF_W1T);
  __bf16* w2t = (__bf16*)(smem + OFF_W2T);
  __bf16* hsl = (__bf16*)(smem + OFF_HSLAB);
  __bf16* qbf = (__bf16*)(smem + OFF_QBF);
  __bf16* w3p = (__bf16*)(smem + OFF_W3P);
  float*  b1s = (float*)(smem + OFF_B1);
  float*  b2s = (float*)(smem + OFF_B2);
  float*  sc  = (float*)(smem + OFF_SC);
  float*  red = (float*)(smem + OFF_RED);
  float*  gv  = (float*)(smem + OFF_G);
  float*  part = (float*)(smem + OFF_PART);

  const int tid    = threadIdx.x;
  const int b      = blockIdx.x;
  const int lane   = tid & 31;
  const int wave   = tid >> 5;
  const int lane15 = lane & 15;
  const int rowhi  = (lane < 16) ? 0 : 8;   // C-row offset == A-frag K low offset
  const int kb16   = (lane < 16) ? 0 : 8;   // B-frag K offset

  // ---- zero padded LDS regions (so K/N padding contributes exactly 0) ----
  for (int i = tid; i < 48 * HSTR; i += 256) w2t[i] = (__bf16)0.0f;
  for (int i = tid; i < 96;        i += 256) w3p[i] = (__bf16)0.0f;
  for (int i = tid; i < 48;        i += 256) b2s[i] = 0.0f;
  for (int i = tid; i < 128 * 16;  i += 256) {       // h-slab cols 80..95
    int r = i >> 4, c = 80 + (i & 15);
    hsl[r * HSTR + c] = (__bf16)0.0f;
  }
  __syncthreads();

  // ---- stage q, key[b], W1^T, W2^T, W3, biases into LDS (f32 -> bf16) ----
  if (tid < D) qbf[tid] = (__bf16)query[(size_t)b * D + tid];
  for (int i = tid; i < TP * D; i += 256) {
    int t = i >> 6, d = i & 63;
    float v = (t < T) ? key[((size_t)b * T + t) * D + d] : 0.0f;
    kbf[t * KSTR + d] = (__bf16)v;
  }
  for (int i = tid; i < 256 * H1; i += 256) {        // W1 is (256 x 80) row-major
    int f = i / H1, n = i - f * H1;
    w1t[n * W1STR + f] = (__bf16)W1[i];
  }
  for (int i = tid; i < H1 * H2; i += 256) {         // W2 is (80 x 40)
    int h = i / H2, g = i - h * H2;
    w2t[g * HSTR + h] = (__bf16)W2[i];
  }
  if (tid < H2) w3p[tid] = (__bf16)W3[tid];
  if (tid < H1) b1s[tid] = b1[tid];
  if (tid < H2) b2s[tid] = b2[tid];
  __syncthreads();

  const int   mlen = mask[b];
  const float b3v  = b3[0];

  // q-region A fragments are row-constant: build once per wave.
  v16bf qfr[2];
#pragma unroll
  for (int c0w = 0; c0w < 2; ++c0w)
    qfr[c0w] = frag_gap((const char*)qbf + (c0w * 32 + rowhi) * 2);

  // W3 B-fragments: only column n==0 carries values, other lanes zero.
  v16bf w3fr[3];
#pragma unroll
  for (int kt = 0; kt < 3; ++kt) {
    if (lane15 == 0) {
      w3fr[kt] = frag_seq((const char*)w3p + (kt * 32 + kb16) * 2);
    } else {
      v16bf z;
#pragma unroll
      for (int i = 0; i < 16; ++i) z[i] = (__bf16)0.0f;
      w3fr[kt] = z;
    }
  }

  v8f zero8 = {0.f, 0.f, 0.f, 0.f, 0.f, 0.f, 0.f, 0.f};

  // ---- per-wave M tiles: fused GEMM1 -> GEMM2 -> GEMM3 (tile-local LDS) ----
  for (int mt = wave; mt < MT; mt += 8) {
    const int trow = mt * 16 + lane15;
    __bf16* hrow = hsl + (wave * 16) * HSTR;

    // GEMM1: din(16x256) x W1(256x80).  din regions: [q | k | q-k | q*k]
    v8f acc1[5];
#pragma unroll
    for (int nt = 0; nt < 5; ++nt) acc1[nt] = zero8;

#pragma unroll
    for (int c0w = 0; c0w < 2; ++c0w) {
      const int c0 = c0w * 32;
      v16bf kf = frag_gap((const char*)kbf + (trow * KSTR + c0 + rowhi) * 2);
      v16bf af[4];
      af[0] = qfr[c0w];
      af[1] = kf;
      af[2] = frag_sub(qfr[c0w], kf);
      af[3] = frag_mul(qfr[c0w], kf);
#pragma unroll
      for (int r = 0; r < 4; ++r) {
        const int kt = r * 2 + c0w;
#pragma unroll
        for (int nt = 0; nt < 5; ++nt) {
          v16bf bf = frag_seq((const char*)w1t +
                              ((nt * 16 + lane15) * W1STR + kt * 32 + kb16) * 2);
          acc1[nt] = __builtin_amdgcn_wmma_f32_16x16x32_bf16(
              false, af[r], false, bf, (short)0, acc1[nt], false, false);
        }
      }
    }
    // bias + sigmoid -> h1 (bf16) into this wave's slab rows
#pragma unroll
    for (int nt = 0; nt < 5; ++nt) {
      const int n = nt * 16 + lane15;
      const float bb = b1s[n];
#pragma unroll
      for (int r = 0; r < 8; ++r) {
        float s = sigmoidf_fast(acc1[nt][r] + bb);
        hrow[(r + rowhi) * HSTR + n] = (__bf16)s;
      }
    }

    // GEMM2: h1(16x96) x W2(96x48)  (K,N zero-padded)
    v8f acc2[3];
#pragma unroll
    for (int nt = 0; nt < 3; ++nt) acc2[nt] = zero8;
#pragma unroll
    for (int kt = 0; kt < 3; ++kt) {
      v16bf a2 = frag_gap((const char*)hrow + (lane15 * HSTR + kt * 32 + rowhi) * 2);
#pragma unroll
      for (int nt = 0; nt < 3; ++nt) {
        v16bf bf = frag_seq((const char*)w2t +
                            ((nt * 16 + lane15) * HSTR + kt * 32 + kb16) * 2);
        acc2[nt] = __builtin_amdgcn_wmma_f32_16x16x32_bf16(
            false, a2, false, bf, (short)0, acc2[nt], false, false);
      }
    }
    // bias + sigmoid -> h2 into slab cols 0..47 (cols 48..95 stay finite/zero)
#pragma unroll
    for (int nt = 0; nt < 3; ++nt) {
      const int n = nt * 16 + lane15;
      const float bb = b2s[n];
#pragma unroll
      for (int r = 0; r < 8; ++r) {
        float s = sigmoidf_fast(acc2[nt][r] + bb);
        hrow[(r + rowhi) * HSTR + n] = (__bf16)s;
      }
    }

    // GEMM3: h2(16x96) x W3(96x16, only column 0 nonzero)
    v8f acc3 = zero8;
#pragma unroll
    for (int kt = 0; kt < 3; ++kt) {
      v16bf a3 = frag_gap((const char*)hrow + (lane15 * HSTR + kt * 32 + rowhi) * 2);
      acc3 = __builtin_amdgcn_wmma_f32_16x16x32_bf16(
          false, a3, false, w3fr[kt], (short)0, acc3, false, false);
    }
    if (lane15 == 0) {  // lanes 0 and 16 hold column n==0 of C
#pragma unroll
      for (int r = 0; r < 8; ++r) {
        int t = mt * 16 + r + rowhi;
        float s = acc3[r] + b3v;
        s = (t < mlen) ? s * 0.125f : NEG_INF * 0.125f;
        sc[t] = s;
      }
    }
  }
  __syncthreads();

  // ---- softmax over 208 padded scores (wave shfl + LDS reduce) ----
  float s0 = (tid < TP) ? sc[tid] : (NEG_INF * 0.125f);
  float m = s0;
#pragma unroll
  for (int off = 16; off; off >>= 1) m = fmaxf(m, __shfl_xor(m, off, 32));
  if (lane == 0) red[wave] = m;
  __syncthreads();
  if (tid == 0) {
    float mm = red[0];
    for (int i = 1; i < 8; ++i) mm = fmaxf(mm, red[i]);
    gv[0] = mm;
  }
  __syncthreads();
  const float gm = gv[0];
  float e = (tid < TP) ? __expf(s0 - gm) : 0.0f;
  float su = e;
#pragma unroll
  for (int off = 16; off; off >>= 1) su += __shfl_xor(su, off, 32);
  if (lane == 0) red[wave] = su;
  __syncthreads();
  if (tid == 0) {
    float ss = red[0];
    for (int i = 1; i < 8; ++i) ss += red[i];
    gv[1] = ss;
  }
  __syncthreads();
  const float inv = __fdividef(1.0f, gv[1]);
  if (tid < TP) sc[tid] = e * inv;
  __syncthreads();

  // ---- out[b,:] = attn . key[b]  (fp32 from global; 4 t-chunks x 64 d) ----
  const int d  = tid & 63;
  const int ch = tid >> 6;
  const float* krow = key + (size_t)b * T * D;
  float acc = 0.0f;
  for (int t = ch * 50; t < ch * 50 + 50; ++t)
    acc += sc[t] * krow[t * D + d];
  part[tid] = acc;
  __syncthreads();
  if (tid < 64)
    out[(size_t)b * D + tid] =
        part[tid] + part[64 + tid] + part[128 + tid] + part[192 + tid];
}

extern "C" void kernel_launch(void* const* d_in, const int* in_sizes, int n_in,
                              void* d_out, int out_size, void* d_ws, size_t ws_size,
                              hipStream_t stream) {
  (void)n_in; (void)out_size; (void)d_ws; (void)ws_size;
  const float* query = (const float*)d_in[0];
  const float* key   = (const float*)d_in[1];
  const int*   mask  = (const int*)d_in[2];
  const float* W1    = (const float*)d_in[3];
  const float* b1    = (const float*)d_in[4];
  const float* W2    = (const float*)d_in[5];
  const float* b2    = (const float*)d_in[6];
  const float* W3    = (const float*)d_in[7];
  const float* b3    = (const float*)d_in[8];
  float* out = (float*)d_out;
  const int B = in_sizes[0] / D;

  (void)hipFuncSetAttribute((const void*)din_attn_kernel,
                            hipFuncAttributeMaxDynamicSharedMemorySize,
                            SMEM_BYTES);
  din_attn_kernel<<<B, 256, SMEM_BYTES, stream>>>(
      query, key, mask, W1, b1, W2, b2, W3, b3, out);
}
